// MultiheadAttention_90134183674046
// MI455X (gfx1250) — matrix operand order repaired and checked by reading, not compiled, not run on hardware
//
#include <hip/hip_runtime.h>

// ---------------------------------------------------------------------------
// MultiheadAttention forward for MI455X (gfx1250, wave32, WMMA bf16 path)
//   B=4, S=2048, D=1024, H=16, Hd=64
// Pipeline (all in d_ws):
//   1) fp32 -> bf16 conversion of activations and weights
//   2) Q/K/V projections: bf16 WMMA GEMM with double-buffered LDS staging
//      (async global->LDS copies when available); Q,K -> [B,H,S,Hd],
//      V -> transposed [B,H,Hd,S]
//   3) flash-style attention per 16-query tile: S^T = K*Q^T via WMMA,
//      online softmax in fp32 (xor-16 ds_swizzle row combine), O^T = V^T*P^T
//      via WMMA, cross-iteration K/V register prefetch; O -> bf16 [B,S,D]
//   4) output projection: same WMMA GEMM, fp32 result + bias -> d_out
// ---------------------------------------------------------------------------

typedef __bf16 bf16_t;
typedef __attribute__((ext_vector_type(16))) __bf16 v16bf;
typedef __attribute__((ext_vector_type(8)))  __bf16 v8bf;
typedef __attribute__((ext_vector_type(8)))  float  v8f;
typedef int v4i_vec __attribute__((vector_size(16)));

#define SEQ    2048
#define DMODEL 1024
#define NHEAD  16
#define HDIM   64
#define NBATCH 4
#define MROWS  (NBATCH * SEQ)   // 8192

// ---- gfx1250 async global->LDS path (guarded; falls back to ld+ds_store) --

#if __has_builtin(__builtin_amdgcn_global_load_async_to_lds_b128) && \
    __has_builtin(__builtin_amdgcn_s_wait_asynccnt)
#define USE_ASYNC_LDS 1
#else
#define USE_ASYNC_LDS 0
#endif

__device__ __forceinline__ void async_copy16(const bf16_t* __restrict__ g,
                                             bf16_t* __restrict__ l) {
#if USE_ASYNC_LDS
  // builtin parameter 0 is '__device__ v4i*' (addrspace(1) int4 pointer);
  // LDS side correspondingly addrspace(3) int4 pointer; then offset, cpol.
  __builtin_amdgcn_global_load_async_to_lds_b128(
      (__attribute__((address_space(1))) v4i_vec*)g,
      (__attribute__((address_space(3))) v4i_vec*)l,
      /*offset=*/0, /*cpol=*/0);
#else
  *reinterpret_cast<v8bf*>(l) = *reinterpret_cast<const v8bf*>(g);
#endif
}

__device__ __forceinline__ void async_join() {
#if USE_ASYNC_LDS
  __builtin_amdgcn_s_wait_asynccnt(0);
#endif
  __syncthreads();
}

// ---- WMMA helpers ---------------------------------------------------------

__device__ __forceinline__ v8f wmma_bf16(v16bf a, v16bf b, v8f c) {
  // D = A(16x32 bf16) * B(32x16 bf16) + C(16x16 f32)
  return __builtin_amdgcn_wmma_f32_16x16x32_bf16(
      /*neg_a=*/false, a, /*neg_b=*/false, b,
      /*c_mod=*/(short)0, c, /*reuse_a=*/false, /*reuse_b=*/false);
}

// A-matrix fragment: 16(M) x 32(K), source row-major with leading dim `ld`
// lane<16: M=lane,   K = 0..15
// lane>=16:M=lane-16,K = 16..31
__device__ __forceinline__ v16bf load_frag_a(const bf16_t* __restrict__ base,
                                             int ld, int lane) {
  const int r   = lane & 15;
  const int sel = (lane >> 4) & 1;
  const bf16_t* p = base + (size_t)r * ld + sel * 16;
  union { v16bf v; v8bf h[2]; } u;
  u.h[0] = *reinterpret_cast<const v8bf*>(p);       // K = sel*16 + 0..7
  u.h[1] = *reinterpret_cast<const v8bf*>(p + 8);   // K = sel*16 + 8..15
  return u.v;
}

// B-matrix fragment: 32(K) x 16(N), sourced from B^T row-major (rows are the
// N-columns, K-contiguous) with leading dim `ld`.
// lane<16: N=lane, K = 0..15 ; lane>=16: N=lane-16, K = 16..31
__device__ __forceinline__ v16bf load_frag_b(const bf16_t* __restrict__ base,
                                             int ld, int lane) {
  const int r   = lane & 15;
  const int sel = (lane >> 4) & 1;
  const bf16_t* p = base + (size_t)r * ld + sel * 16;
  union { v16bf v; v8bf h[2]; } u;
  u.h[0] = *reinterpret_cast<const v8bf*>(p);
  u.h[1] = *reinterpret_cast<const v8bf*>(p + 8);
  return u.v;
}

// xor-16 lane swap (SWAPX16): offset[15]=0, xor_mask=0x10, and_mask=0x1f
__device__ __forceinline__ float xswap16(float x) {
  return __int_as_float(__builtin_amdgcn_ds_swizzle(__float_as_int(x), 0x401F));
}

// ---- fp32 -> bf16 streaming conversion ------------------------------------

__global__ void cvt_f32_bf16(const float* __restrict__ in,
                             bf16_t* __restrict__ out, int n4) {
  int i = blockIdx.x * blockDim.x + threadIdx.x;
  if (i < n4) {
    float4 f = reinterpret_cast<const float4*>(in)[i];
    union { bf16_t b[4]; unsigned long long u; } p;
    p.b[0] = (bf16_t)f.x; p.b[1] = (bf16_t)f.y;
    p.b[2] = (bf16_t)f.z; p.b[3] = (bf16_t)f.w;
    reinterpret_cast<unsigned long long*>(out)[i] = p.u;
  }
}

// ---- projection GEMM: C[m,n] = sum_k A[m,k] * W[n,k] + bias[n] ------------
// Block tile: 128(M) x 64(N), 8 waves each own a 16x64 slice; K chunks of 32
// staged through double-buffered LDS with async copies.
// mode 0: bf16 out, head layout [B,H,S,Hd]
// mode 1: bf16 out, transposed head layout [B,H,Hd,S]   (for V)
// mode 2: fp32 out, row-major [MROWS, DMODEL]           (final projection)

__global__ __launch_bounds__(256)
void gemm_bf16_kernel(const bf16_t* __restrict__ A,   // [MROWS, DMODEL]
                      const bf16_t* __restrict__ W,   // [DMODEL, DMODEL]
                      const float*  __restrict__ bias,
                      bf16_t* __restrict__ outB,
                      float*  __restrict__ outF,
                      int mode) {
  __shared__ bf16_t lA[2][128 * 32];   // 8 KB per buffer
  __shared__ bf16_t lW[2][64 * 32];    // 4 KB per buffer

  const int tid  = threadIdx.x;
  const int lane = tid & 31;
  const int wave = tid >> 5;                    // 8 waves
  const int m0 = blockIdx.x * 128;
  const int n0 = blockIdx.y * HDIM;

  // stage K-chunk [k0, k0+32) into LDS buffer `buf` (3 x 16B per thread)
  auto stage = [&](int buf, int k0) {
#pragma unroll
    for (int p = 0; p < 2; ++p) {               // A: 512 x 16B parts
      const int part = tid * 2 + p;
      const int row  = part >> 2;               // 0..127
      const int c    = part & 3;                // 16B chunk within 64B row
      async_copy16(A + (size_t)(m0 + row) * DMODEL + k0 + c * 8,
                   &lA[buf][row * 32 + c * 8]);
    }
    {                                           // W: 256 x 16B parts
      const int row = tid >> 2;                 // 0..63
      const int c   = tid & 3;
      async_copy16(W + (size_t)(n0 + row) * DMODEL + k0 + c * 8,
                   &lW[buf][row * 32 + c * 8]);
    }
  };

  v8f acc[4];
#pragma unroll
  for (int t = 0; t < 4; ++t)
#pragma unroll
    for (int i = 0; i < 8; ++i) acc[t][i] = 0.0f;

  stage(0, 0);
  for (int k0 = 0, step = 0; k0 < DMODEL; k0 += 32, ++step) {
    const int cur = step & 1;
    async_join();                       // buf[cur] ready; prev reads done
    if (k0 + 32 < DMODEL) {
      stage(cur ^ 1, k0 + 32);          // overlap next chunk with compute
      if (k0 + 64 < DMODEL) {           // L2 prefetch two chunks ahead
        __builtin_prefetch(A + (size_t)(m0 + (tid >> 1)) * DMODEL + k0 + 64,
                           0, 0);
      }
    }
    const v16bf a = load_frag_a(&lA[cur][(wave * 16) * 32], 32, lane);
#pragma unroll
    for (int t = 0; t < 4; ++t) {
      const v16bf b = load_frag_b(&lW[cur][(t * 16) * 32], 32, lane);
      acc[t] = wmma_bf16(a, b, acc[t]);
    }
  }

  const int  mrow = m0 + wave * 16;
  const int  col  = lane & 15;
  const int  sel  = (lane >> 4) & 1;
  const int  bi   = mrow / SEQ;
  const int  srow = mrow % SEQ;
  const int  h    = blockIdx.y;

#pragma unroll
  for (int t = 0; t < 4; ++t) {
    const int n  = n0 + t * 16 + col;
    const float bv = bias[n];
    if (mode == 2) {
#pragma unroll
      for (int i = 0; i < 8; ++i) {
        const int m = mrow + sel * 8 + i;
        outF[(size_t)m * DMODEL + n] = acc[t][i] + bv;
      }
    } else if (mode == 0) {
      const int d = t * 16 + col;
      bf16_t* dst = outB + ((size_t)bi * NHEAD + h) * SEQ * HDIM + d;
#pragma unroll
      for (int i = 0; i < 8; ++i) {
        const int s = srow + sel * 8 + i;
        dst[(size_t)s * HDIM] = (bf16_t)(acc[t][i] + bv);
      }
    } else {  // mode 1: V transposed [B,H,Hd,S] -> contiguous 16B store
      const int d = t * 16 + col;
      v8bf pk;
#pragma unroll
      for (int i = 0; i < 8; ++i) pk[i] = (bf16_t)(acc[t][i] + bv);
      bf16_t* dst = outB + (((size_t)bi * NHEAD + h) * HDIM + d) * SEQ +
                    srow + sel * 8;
      *reinterpret_cast<v8bf*>(dst) = pk;
    }
  }
}

// ---- flash attention per 16-query tile ------------------------------------

__global__ __launch_bounds__(128)
void attn_kernel(const bf16_t* __restrict__ Q,    // [B,H,S,Hd]
                 const bf16_t* __restrict__ K,    // [B,H,S,Hd]
                 const bf16_t* __restrict__ VT,   // [B,H,Hd,S]
                 bf16_t* __restrict__ O) {        // [B,S,D]
  const int lane = threadIdx.x & 31;
  const int wave = threadIdx.x >> 5;               // 4 waves/block
  const int h = blockIdx.y;
  const int b = blockIdx.z;
  const int q0 = (blockIdx.x * 4 + wave) * 16;

  const bf16_t* Qb = Q + ((size_t)b * NHEAD + h) * SEQ * HDIM;
  const bf16_t* Kb = K + ((size_t)b * NHEAD + h) * SEQ * HDIM;
  const bf16_t* Vb = VT + ((size_t)b * NHEAD + h) * HDIM * SEQ;

  // Q tile as B-fragments (constant over the key loop)
  const v16bf qf0 = load_frag_b(Qb + (size_t)q0 * HDIM + 0, HDIM, lane);
  const v16bf qf1 = load_frag_b(Qb + (size_t)q0 * HDIM + 32, HDIM, lane);

  auto load_k = [&](v16bf* kd, int kt) {
    kd[0] = load_frag_a(Kb + (size_t)kt * HDIM, HDIM, lane);
    kd[1] = load_frag_a(Kb + (size_t)kt * HDIM + 32, HDIM, lane);
    kd[2] = load_frag_a(Kb + (size_t)(kt + 16) * HDIM, HDIM, lane);
    kd[3] = load_frag_a(Kb + (size_t)(kt + 16) * HDIM + 32, HDIM, lane);
  };
  auto load_v = [&](v16bf* vd, int kt) {
#pragma unroll
    for (int t = 0; t < 4; ++t)
      vd[t] = load_frag_a(Vb + (size_t)(t * 16) * SEQ + kt, SEQ, lane);
  };

  v8f o[4];
#pragma unroll
  for (int t = 0; t < 4; ++t)
#pragma unroll
    for (int i = 0; i < 8; ++i) o[t][i] = 0.0f;

  float m_run = -3.0e38f;
  float l_run = 0.0f;
  const float scale = 0.125f;  // 1/sqrt(64)

  v16bf ka[4], va[4];
  load_k(ka, 0);
  load_v(va, 0);

  for (int kt = 0; kt < SEQ; kt += 32) {
    // prefetch next iteration's fragments; they complete behind the softmax
    v16bf kn[4], vn[4];
    const int ktn = (kt + 32 < SEQ) ? (kt + 32) : kt;
    load_k(kn, ktn);
    load_v(vn, ktn);

    // S^T tiles: s0 = keys kt..kt+15, s1 = kt+16..kt+31 (rows=key, cols=q)
    v8f s0, s1;
#pragma unroll
    for (int i = 0; i < 8; ++i) { s0[i] = 0.0f; s1[i] = 0.0f; }
    s0 = wmma_bf16(ka[0], qf0, s0);
    s0 = wmma_bf16(ka[1], qf1, s0);
    s1 = wmma_bf16(ka[2], qf0, s1);
    s1 = wmma_bf16(ka[3], qf1, s1);

    // online softmax; lane owns query q=lane&15; xor-16 partner holds the
    // complementary 8 key-rows of each tile
    float tmax = -3.0e38f;
#pragma unroll
    for (int i = 0; i < 8; ++i) {
      s0[i] *= scale;
      s1[i] *= scale;
      tmax = fmaxf(tmax, fmaxf(s0[i], s1[i]));
    }
    tmax = fmaxf(tmax, xswap16(tmax));
    const float m_new = fmaxf(m_run, tmax);
    const float corr = __expf(m_run - m_new);

    float psum = 0.0f;
    float p0[8], p1[8];
#pragma unroll
    for (int i = 0; i < 8; ++i) {
      p0[i] = __expf(s0[i] - m_new);
      p1[i] = __expf(s1[i] - m_new);
      psum += p0[i] + p1[i];
    }
    l_run = l_run * corr + (psum + xswap16(psum));
    m_run = m_new;

    // build P^T B-fragment for keys kt..kt+31:
    //  lane<16 (q=lane):     k 0..15  -> tile0 rows 0..7 (own) + 8..15 (partner)
    //  lane>=16 (q=lane-16): k 16..31 -> tile1 rows 0..7 (partner) + 8..15 (own)
    const int sel = (lane >> 4) & 1;
    v16bf pb;
#pragma unroll
    for (int i = 0; i < 8; ++i) {
      const float x0 = xswap16(p0[i]);
      const float x1 = xswap16(p1[i]);
      pb[i]     = (bf16_t)(sel ? x1 : p0[i]);
      pb[i + 8] = (bf16_t)(sel ? p1[i] : x0);
    }

    // O^T[d,q] += V^T[d,k] * P^T[k,q]
#pragma unroll
    for (int t = 0; t < 4; ++t) {
#pragma unroll
      for (int i = 0; i < 8; ++i) o[t][i] *= corr;
      o[t] = wmma_bf16(va[t], pb, o[t]);
    }

#pragma unroll
    for (int t = 0; t < 4; ++t) { ka[t] = kn[t]; va[t] = vn[t]; }
  }

  // finalize: divide by softmax denominator, store O[b, q, h*64 + d] (bf16)
  const float rl = 1.0f / l_run;
  const int col = lane & 15;
  const int sel = (lane >> 4) & 1;
  const int q = q0 + col;
  bf16_t* dst = O + ((size_t)b * SEQ + q) * DMODEL + h * HDIM;
#pragma unroll
  for (int t = 0; t < 4; ++t) {
    v8bf pk;
#pragma unroll
    for (int i = 0; i < 8; ++i) pk[i] = (bf16_t)(o[t][i] * rl);
    *reinterpret_cast<v8bf*>(dst + t * 16 + sel * 8) = pk;
  }
}

// ---- host launcher --------------------------------------------------------

extern "C" void kernel_launch(void* const* d_in, const int* in_sizes, int n_in,
                              void* d_out, int out_size, void* d_ws,
                              size_t ws_size, hipStream_t stream) {
  const float* query = (const float*)d_in[0];
  const float* key_  = (const float*)d_in[1];
  const float* value = (const float*)d_in[2];
  const float* Wq = (const float*)d_in[3];
  const float* bq = (const float*)d_in[4];
  const float* Wk = (const float*)d_in[5];
  const float* bk = (const float*)d_in[6];
  const float* Wv = (const float*)d_in[7];
  const float* bv = (const float*)d_in[8];
  const float* Wo = (const float*)d_in[9];
  const float* bo = (const float*)d_in[10];
  float* out = (float*)d_out;

  bf16_t* ws = (bf16_t*)d_ws;
  const size_t SZ_X = (size_t)MROWS * DMODEL;    // 8.39M elems
  const size_t SZ_W = (size_t)DMODEL * DMODEL;   // 1.05M elems
  bf16_t* XQ  = ws;             bf16_t* XK  = XQ + SZ_X;
  bf16_t* XV  = XK + SZ_X;      bf16_t* WQb = XV + SZ_X;
  bf16_t* WKb = WQb + SZ_W;     bf16_t* WVb = WKb + SZ_W;
  bf16_t* WOb = WVb + SZ_W;     bf16_t* Qh  = WOb + SZ_W;
  bf16_t* Kh  = Qh + SZ_X;      bf16_t* Vt  = Kh + SZ_X;
  bf16_t* Oc  = Vt + SZ_X;

  auto cvt = [&](const float* src, bf16_t* dst, size_t n) {
    int n4 = (int)(n / 4);
    cvt_f32_bf16<<<(n4 + 255) / 256, 256, 0, stream>>>(src, dst, n4);
  };
  cvt(query, XQ, SZ_X);
  cvt(key_,  XK, SZ_X);
  cvt(value, XV, SZ_X);
  cvt(Wq, WQb, SZ_W);
  cvt(Wk, WKb, SZ_W);
  cvt(Wv, WVb, SZ_W);
  cvt(Wo, WOb, SZ_W);

  dim3 gG(MROWS / 128, DMODEL / HDIM, 1);
  gemm_bf16_kernel<<<gG, 256, 0, stream>>>(XQ, WQb, bq, Qh, nullptr, 0);
  gemm_bf16_kernel<<<gG, 256, 0, stream>>>(XK, WKb, bk, Kh, nullptr, 0);
  gemm_bf16_kernel<<<gG, 256, 0, stream>>>(XV, WVb, bv, Vt, nullptr, 1);

  dim3 gA(SEQ / 64, NHEAD, NBATCH);
  attn_kernel<<<gA, 128, 0, stream>>>(Qh, Kh, Vt, Oc);

  gemm_bf16_kernel<<<gG, 256, 0, stream>>>(Oc, WOb, bo, nullptr, out, 2);
}
